// myQTransitionBlock_36498632081877
// MI455X (gfx1250) — 8-bit matrix operand order repaired and checked by reading, compiled, not run on hardware
//
#include <hip/hip_runtime.h>

#define C_ 256
#define S_ 4
#define N_ 16
#define F_ 64
#define T_ 256
#define EPS_ 1e-5f

typedef __attribute__((ext_vector_type(8))) int v8i;

__device__ __forceinline__ float qfix(float x, float scale, float inv) {
  float r = rintf(x * scale);
  r = fminf(fmaxf(r, -128.0f), 127.0f);
  return r * inv;
}
__device__ __forceinline__ int qfix_i(float x, float scale) {
  float r = rintf(x * scale);
  r = fminf(fmaxf(r, -128.0f), 127.0f);
  return (int)r;
}

__global__ void k_stage1(const float* __restrict__ x,
                         const float* __restrict__ w_freq,
                         const float* __restrict__ sg, const float* __restrict__ sb,
                         const float* __restrict__ sm, const float* __restrict__ sv,
                         signed char* __restrict__ aux,
                         signed char* __restrict__ meanq) {
  const int nc = blockIdx.x;
  const int c  = nc & (C_ - 1);
  const int t  = threadIdx.x;

  const float w0 = qfix(w_freq[c * 3 + 0], 128.f, 1.f / 128.f);
  const float w1 = qfix(w_freq[c * 3 + 1], 128.f, 1.f / 128.f);
  const float w2 = qfix(w_freq[c * 3 + 2], 128.f, 1.f / 128.f);

  float ms[S_], bs[S_];
#pragma unroll
  for (int s = 0; s < S_; ++s) {
    const int i  = c * S_ + s;
    const float m = qfix(sg[i] / sqrtf(sv[i] + EPS_), 128.f, 1.f / 128.f);
    ms[s] = m;
    bs[s] = qfix(sb[i] - sm[i] * m, 16.f, 1.f / 16.f);
  }

  const float*  xp = x   + (size_t)nc * (F_ * T_) + t;
  signed char*  ap = aux + (size_t)nc * (F_ * T_) + t;

  float xm = 0.f;
  float x0 = xp[0];
  int   sumq = 0;
#pragma unroll 4
  for (int f = 0; f < F_; ++f) {
    const float xn = (f < F_ - 1) ? xp[(f + 1) * T_] : 0.f;
    const float y  = w0 * xm + w1 * x0 + w2 * xn;
    const float a  = qfix(y, 16.f, 1.f / 16.f);
    const float z  = a * ms[f >> 4] + bs[f >> 4];
    const int   q  = qfix_i(z, 16.f);
    ap[f * T_] = (signed char)q;
    sumq += q;
    xm = x0; x0 = xn;
  }
  const int mq = qfix_i((float)sumq * (1.0f / 1024.0f), 16.f);
  meanq[(size_t)nc * T_ + t] = (signed char)mq;
}

__global__ void k_stage2(const signed char* __restrict__ meanq,
                         const float* __restrict__ w_temp,
                         const float* __restrict__ g2, const float* __restrict__ b2,
                         const float* __restrict__ mu2, const float* __restrict__ v2,
                         signed char* __restrict__ gact) {
  const int n = blockIdx.x >> 8;
  const int t = blockIdx.x & 255;
  const int c = threadIdx.x;

  const float wt0 = qfix(w_temp[c * 3 + 0], 128.f, 1.f / 128.f);
  const float wt1 = qfix(w_temp[c * 3 + 1], 128.f, 1.f / 128.f);
  const float wt2 = qfix(w_temp[c * 3 + 2], 128.f, 1.f / 128.f);
  const float m2  = qfix(g2[c] / sqrtf(v2[c] + EPS_), 128.f, 1.f / 128.f);
  const float bb2 = qfix(b2[c] - mu2[c] * m2, 16.f, 1.f / 16.f);

  const signed char* mp = meanq + (size_t)(n * C_ + c) * T_;
  const float vm = (t > 0)      ? (float)mp[t - 1] : 0.f;
  const float v0 = (float)mp[t];
  const float vp = (t < T_ - 1) ? (float)mp[t + 1] : 0.f;

  const float u  = (wt0 * vm + wt1 * v0 + wt2 * vp) * (1.f / 16.f);
  const int   uq = qfix_i(u, 16.f);
  const float v  = (float)uq * (1.f / 16.f) * m2 + bb2;
  int vq = qfix_i(v, 16.f);
  if (vq < 0) vq = 0;
  gact[(size_t)(n * T_ + t) * C_ + c] = (signed char)vq;
}

__global__ void k_wq(const float* __restrict__ w, signed char* __restrict__ wq) {
  const int i = blockIdx.x * blockDim.x + threadIdx.x;
  wq[i] = (signed char)qfix_i(w[i], 128.f);
}

__global__ void k_gemm(const signed char* __restrict__ gact,
                       const signed char* __restrict__ wq,
                       signed char* __restrict__ dq) {
  const int lane   = threadIdx.x & 31;
  const int tile   = blockIdx.x * 8 + (threadIdx.x >> 5);
  const int o_base = (tile & 15) << 4;
  const int m_base = (tile >> 4) << 4;
  const int half   = lane >> 4;
  const int l16    = lane & 15;
  const int mrow   = m_base + l16;
  const int ocol   = o_base + l16;

  const signed char* arow = gact + (size_t)mrow * C_;
  const signed char* brow = wq   + (size_t)ocol * C_;

  v8i acc = {};
#pragma unroll
  for (int ks = 0; ks < 4; ++ks) {
    const int kb = ks * 64;
    const signed char* ap = arow + kb + half * 16;
    const int4 a0 = *(const int4*)(ap);
    const int4 a1 = *(const int4*)(ap + 32);
    v8i A;
    A[0] = a0.x; A[1] = a0.y; A[2] = a0.z; A[3] = a0.w;
    A[4] = a1.x; A[5] = a1.y; A[6] = a1.z; A[7] = a1.w;
    const signed char* bp = brow + kb + half * 16;
    const int4 b0 = *(const int4*)(bp);
    const int4 b1 = *(const int4*)(bp + 32);
    v8i B;
    B[0] = b0.x; B[1] = b0.y; B[2] = b0.z; B[3] = b0.w;
    B[4] = b1.x; B[5] = b1.y; B[6] = b1.z; B[7] = b1.w;
    acc = __builtin_amdgcn_wmma_i32_16x16x64_iu8(true, A, true, B, acc,
                                                 false, false);
  }

  const int m0 = m_base + half * 8;
#pragma unroll
  for (int g = 0; g < 8; ++g) {
    const int m  = m0 + g;
    const float val = (float)acc[g] * (1.0f / 2048.0f);
    const int dv = qfix_i(val, 16.f);
    const int n = m >> 8, t = m & 255;
    dq[(size_t)(n * C_ + ocol) * T_ + t] = (signed char)dv;
  }
}

__global__ void k_epilogue(const signed char* __restrict__ aux,
                           const signed char* __restrict__ dq,
                           float* __restrict__ out) {
  const size_t tid  = (size_t)blockIdx.x * blockDim.x + threadIdx.x;
  const size_t base = tid * 4;
  const int    t    = (int)(base & 255);
  const size_t row  = base >> 8;
  const size_t nc   = row >> 6;

  const char4 qa = *(const char4*)(aux + base);
  const char4 qd = *(const char4*)(dq + nc * T_ + t);

  float4 r;
  int s;
  s = qa.x + qd.x; s = s > 127 ? 127 : (s < -128 ? -128 : s); r.x = s > 0 ? (float)s * (1.f/16.f) : 0.f;
  s = qa.y + qd.y; s = s > 127 ? 127 : (s < -128 ? -128 : s); r.y = s > 0 ? (float)s * (1.f/16.f) : 0.f;
  s = qa.z + qd.z; s = s > 127 ? 127 : (s < -128 ? -128 : s); r.z = s > 0 ? (float)s * (1.f/16.f) : 0.f;
  s = qa.w + qd.w; s = s > 127 ? 127 : (s < -128 ? -128 : s); r.w = s > 0 ? (float)s * (1.f/16.f) : 0.f;
  *(float4*)(out + base) = r;
}

extern "C" void kernel_launch(void* const* d_in, const int* in_sizes, int n_in,
                              void* d_out, int out_size, void* d_ws, size_t ws_size,
                              hipStream_t stream) {
  (void)in_sizes; (void)n_in; (void)out_size; (void)ws_size;
  const float* x      = (const float*)d_in[0];
  const float* w_freq = (const float*)d_in[1];
  const float* sg     = (const float*)d_in[2];
  const float* sb     = (const float*)d_in[3];
  const float* smn    = (const float*)d_in[4];
  const float* sv     = (const float*)d_in[5];
  const float* w_temp = (const float*)d_in[6];
  const float* g2     = (const float*)d_in[7];
  const float* b2     = (const float*)d_in[8];
  const float* mu2    = (const float*)d_in[9];
  const float* v2     = (const float*)d_in[10];
  const float* w1x1   = (const float*)d_in[11];
  float* out = (float*)d_out;

  signed char* ws    = (signed char*)d_ws;
  signed char* aux   = ws;
  signed char* meanq = aux   + (size_t)N_ * C_ * F_ * T_;
  signed char* gact  = meanq + (size_t)N_ * C_ * T_;
  signed char* wq    = gact  + (size_t)N_ * C_ * T_;
  signed char* dq    = wq    + (size_t)C_ * C_;

  k_stage1  <<<N_ * C_, T_, 0, stream>>>(x, w_freq, sg, sb, smn, sv, aux, meanq);
  k_stage2  <<<N_ * T_, C_, 0, stream>>>(meanq, w_temp, g2, b2, mu2, v2, gact);
  k_wq      <<<C_, C_, 0, stream>>>(w1x1, wq);
  k_gemm    <<<(N_ * T_ / 16) * (C_ / 16) / 8, 256, 0, stream>>>(gact, wq, dq);
  k_epilogue<<<(N_ * C_ * F_ * T_) / (4 * 256), 256, 0, stream>>>(aux, dq, out);
}
